// CausalSqueezeExcite1d_14594298872274
// MI455X (gfx1250) — matrix operand order repaired and checked by reading, not compiled, not run on hardware
//
#include <hip/hip_runtime.h>
#include <hip/hip_fp16.h>

// Problem dims (fixed by reference)
#define BB   16
#define CC   512
#define TN   4096
#define HH   32

typedef __attribute__((ext_vector_type(16))) _Float16 v16h;
typedef __attribute__((ext_vector_type(8)))  _Float16 v8h;
typedef __attribute__((ext_vector_type(8)))  float    v8f;

static __device__ __forceinline__ v16h cat8(v8h lo, v8h hi) {
    return __builtin_shufflevector(lo, hi, 0,1,2,3,4,5,6,7,8,9,10,11,12,13,14,15);
}

// ---------------------------------------------------------------------------
// Kernel A: causal EMA scan, x(B,C,T) f32 -> y(B,T,C) f16.
// T split into 4 segments of 1024 with 384-step zero-carry warm-up
// (0.9^384 ~ 3e-18: exact at f32 precision). 256 channels/block; tiles of
// 32 t staged through LDS so global loads are 128B-coalesced per wave and the
// per-channel scan reads LDS at conflict-free pitch 33.
// ---------------------------------------------------------------------------
#define SEG    1024
#define WARM   384
#define CBLK   256
#define TTILE  32

__global__ __launch_bounds__(256) void ema_scan_kernel(
    const float* __restrict__ x, const int* __restrict__ cw,
    _Float16* __restrict__ y)
{
    __shared__ float tile[CBLK * 33];
    const int tid = threadIdx.x;
    const int blk = blockIdx.x;
    const int cg  = blk & 1;          // channel half
    const int s   = (blk >> 1) & 3;   // segment
    const int b   = blk >> 3;         // batch
    const int c0  = cg * CBLK;

    const float a   = 1.0f / (float)cw[0];
    const float oma = 1.0f - a;
    const int tseg   = s * SEG;
    const int tw     = (s == 0) ? 0 : (tseg - WARM);
    const int ntiles = (tseg + SEG - tw) / TTILE;

    const float* xb = x + (size_t)b * CC * TN;
    _Float16*    yb = y + (size_t)b * TN * CC + c0;

    float yv = 0.0f;  // zero carry at warm-up start (exact: decay underflows)
    for (int ti = 0; ti < ntiles; ++ti) {
        const int t0 = tw + ti * TTILE;
        // cooperative coalesced load of CBLK x TTILE tile
        #pragma unroll
        for (int i = 0; i < (CBLK * TTILE) / 256; ++i) {
            int e  = i * 256 + tid;
            int cl = e >> 5;          // /TTILE
            int tl = e & (TTILE - 1);
            tile[cl * 33 + tl] = xb[(size_t)(c0 + cl) * TN + t0 + tl];
        }
        __syncthreads();
        // sequential scan: thread <-> channel, conflict-free LDS reads
        #pragma unroll 4
        for (int tl = 0; tl < TTILE; ++tl) {
            float v = tile[tid * 33 + tl];
            int   t = t0 + tl;
            yv = (t == 0) ? v : fmaf(oma, yv, a * v);
            if (t >= tseg) yb[(size_t)t * CC + tid] = (_Float16)yv;
        }
        __syncthreads();
    }
}

// ---------------------------------------------------------------------------
// Kernel B: fused squeeze/excite GEMMs + sigmoid + gating, all WMMA f16.
// Block = 128 tokens of one batch (8 waves x 16-token M-tile).
//   GEMM1: h(16x32) = relu(Y(16x512) x W1^T(512x32) + b1)   -> 32 wmma/wave
//   GEMM2: g(16x512) = sigmoid(h(16x32) x W2^T(32x512) + b2)-> 32 wmma/wave
//   out[b,c,t] = x[b,c,t] * g[t,c]
// W1/W2 staged to LDS as f16 (padded pitches keep all ds b128 loads 16B
// aligned); h transposed lane-layout via per-wave LDS scratch.
// ---------------------------------------------------------------------------
#define W1P 520   // halves; 1040 B row, 16B multiple
#define W2P 48    // halves; 96 B row
#define HP  48    // halves; 96 B row

__global__ __launch_bounds__(256) void se_wmma_kernel(
    const float* __restrict__ x,  const float* __restrict__ w1,
    const float* __restrict__ b1, const float* __restrict__ w2,
    const float* __restrict__ b2, const _Float16* __restrict__ y,
    float* __restrict__ out)
{
    extern __shared__ char smem[];
    _Float16* w1l = (_Float16*)smem;          // 32  x W1P
    _Float16* w2l = w1l + HH * W1P;           // 512 x W2P
    _Float16* hl  = w2l + CC * W2P;           // 8 waves x 16 x HP

    const int tid  = threadIdx.x;
    const int wave = tid >> 5;
    const int lane = tid & 31;
    const int lrow = lane & 15;
    const int lhi  = lane >> 4;

    // stage weights (f32 -> f16), coalesced
    for (int i = tid; i < HH * CC; i += 256) {
        int r = i >> 9, k = i & 511;
        w1l[r * W1P + k] = (_Float16)w1[i];
    }
    for (int i = tid; i < CC * HH; i += 256) {
        int r = i >> 5, k = i & 31;
        w2l[r * W2P + k] = (_Float16)w2[i];
    }
    __syncthreads();

    const int b     = blockIdx.x >> 5;
    const int t0    = (blockIdx.x & 31) * 128;
    const int twave = t0 + wave * 16;
    const _Float16* yb = y + (size_t)b * TN * CC;

    // ---- GEMM1: accumulate over K=512 in 16 steps of 32 ----
    v8f acc[2] = {};
    #pragma unroll
    for (int kc = 0; kc < CC; kc += 32) {
        // A fragment: lo lanes K 0-7 & 16-23, hi lanes K 8-15 & 24-31
        const _Float16* ap = yb + (size_t)(twave + lrow) * CC + kc + lhi * 8;
        v16h afrag = cat8(*(const v8h*)ap, *(const v8h*)(ap + 16));
        #pragma unroll
        for (int nt = 0; nt < 2; ++nt) {
            // B fragment: lo lanes K 0-7 & 16-23, hi lanes K 8-15 & 24-31
            const _Float16* bp = w1l + (nt * 16 + lrow) * W1P + kc + lhi * 8;
            v16h bfrag = cat8(*(const v8h*)bp, *(const v8h*)(bp + 16));
            acc[nt] = __builtin_amdgcn_wmma_f32_16x16x32_f16(
                false, afrag, false, bfrag, (short)0, acc[nt], false, false);
        }
    }

    // bias + relu, deposit h tile (16 tok x 32 units) to per-wave LDS as f16
    _Float16* hw = hl + wave * 16 * HP;
    #pragma unroll
    for (int nt = 0; nt < 2; ++nt) {
        float bias = b1[nt * 16 + lrow];          // N = lane&15
        #pragma unroll
        for (int v = 0; v < 8; ++v) {             // M = v + 8*lhi
            float hv = fmaxf(acc[nt][v] + bias, 0.0f);
            hw[(v + 8 * lhi) * HP + nt * 16 + lrow] = (_Float16)hv;
        }
    }
    asm volatile("s_wait_dscnt 0" ::: "memory");  // LDS in-order per wave

    // reload h as A fragment (token-row lane layout)
    const _Float16* a2p = hw + lrow * HP + lhi * 8;
    v16h a2 = cat8(*(const v8h*)a2p, *(const v8h*)(a2p + 16));

    const float* xb = x   + (size_t)b * CC * TN;
    float*       ob = out + (size_t)b * CC * TN;
    const int tlane = twave + lhi * 8;            // 8 consecutive t per lane

    // ---- GEMM2 + sigmoid + gate: 32 channel tiles of 16 ----
    for (int ct = 0; ct < 32; ++ct) {
        const _Float16* bp = w2l + (ct * 16 + lrow) * W2P + lhi * 8;
        v16h b2f = cat8(*(const v8h*)bp, *(const v8h*)(bp + 16));
        v8f gz = {};
        gz = __builtin_amdgcn_wmma_f32_16x16x32_f16(
            false, a2, false, b2f, (short)0, gz, false, false);

        const int c = ct * 16 + lrow;
        const float bias2 = b2[c];
        const float4* xp = (const float4*)(xb + (size_t)c * TN + tlane);
        float4*       op = (float4*)(ob + (size_t)c * TN + tlane);
        float4 x0 = xp[0], x1 = xp[1];

        float g[8];
        #pragma unroll
        for (int v = 0; v < 8; ++v) {
            float z = gz[v] + bias2;
            g[v] = 1.0f / (1.0f + __expf(-z));
        }
        float4 o0, o1;
        o0.x = x0.x * g[0]; o0.y = x0.y * g[1]; o0.z = x0.z * g[2]; o0.w = x0.w * g[3];
        o1.x = x1.x * g[4]; o1.y = x1.y * g[5]; o1.z = x1.z * g[6]; o1.w = x1.w * g[7];
        op[0] = o0; op[1] = o1;
    }
}

// ---------------------------------------------------------------------------
extern "C" void kernel_launch(void* const* d_in, const int* in_sizes, int n_in,
                              void* d_out, int out_size, void* d_ws, size_t ws_size,
                              hipStream_t stream)
{
    (void)in_sizes; (void)n_in; (void)out_size; (void)ws_size;
    const float* x  = (const float*)d_in[0];
    const float* w1 = (const float*)d_in[1];
    const float* b1 = (const float*)d_in[2];
    const float* w2 = (const float*)d_in[3];
    const float* b2 = (const float*)d_in[4];
    const int*   cw = (const int*)d_in[5];
    float* out = (float*)d_out;

    // workspace: y in (B,T,C) f16 = 64 MB at offset 0
    _Float16* y = (_Float16*)d_ws;

    // Kernel A: 16 batches x 4 segments x 2 channel-halves = 128 blocks
    ema_scan_kernel<<<dim3(BB * 4 * 2), dim3(256), 0, stream>>>(x, cw, y);

    // Kernel B: (B*T)/128 = 512 token-tile blocks; ~92.5 KB dynamic LDS
    size_t smem = (size_t)(HH * W1P + CC * W2P + 8 * 16 * HP) * sizeof(_Float16);
    (void)hipFuncSetAttribute((const void*)se_wmma_kernel,
                              hipFuncAttributeMaxDynamicSharedMemorySize,
                              (int)smem);
    se_wmma_kernel<<<dim3((BB * TN) / 128), dim3(256), smem, stream>>>(
        x, w1, b1, w2, b2, y, out);
}
